// MultiHeadAttention_83657372991749
// MI455X (gfx1250) — matrix operand order repaired and checked by reading, not compiled, not run on hardware
//
#include <hip/hip_runtime.h>
#include <cstdint>
#include <cstddef>

// ---------------------------------------------------------------------------
// MultiHeadAttention forward for MI455X (gfx1250, wave32, WMMA).
// B=4, S=2048, D=1024, H=16, depth=64.
//
// Pipeline:
//   0) cvt_f32_bf16 x7  : one-shot bf16 conversion of activations + weights
//   1) wmma_gemm  x3    : Q/K/V projections (8192x1024x1024, bf16 in/out)
//   2) flash_attn       : per (b,h) online-softmax attention, WMMA everywhere
//   3) wmma_gemm        : output dense (bf16 in, fp32 out)
// Hot loops contain only b128 staging + v_wmma_f32_16x16x32_bf16.
// ---------------------------------------------------------------------------

typedef unsigned short u16;
typedef __bf16 bf16x16 __attribute__((ext_vector_type(16)));
typedef float  f32x8   __attribute__((ext_vector_type(8)));
typedef int    v4i     __attribute__((ext_vector_type(4)));

#define DEV __device__ __forceinline__

static constexpr int Bb = 4;
static constexpr int S  = 2048;
static constexpr int D  = 1024;
static constexpr int DEP = 64;          // D / H
static constexpr int M_TOK = Bb * S;    // 8192 token rows

// ---- optional CDNA5 async global->LDS path (ASYNCcnt) --------------------
#if defined(__has_builtin)
#  if __has_builtin(__builtin_amdgcn_global_load_async_to_lds_b128) && \
      __has_builtin(__builtin_amdgcn_s_wait_asynccnt)
#    define USE_ASYNC_LDS 1
#  endif
#endif
#ifndef USE_ASYNC_LDS
#  define USE_ASYNC_LDS 0
#endif

DEV void copy16_to_lds(u16* lds_dst, const u16* gsrc) {
#if USE_ASYNC_LDS
  __builtin_amdgcn_global_load_async_to_lds_b128(
      (__attribute__((address_space(1))) v4i*)gsrc,
      (__attribute__((address_space(3))) v4i*)lds_dst, 0, 0);
#else
  *(uint4*)lds_dst = *(const uint4*)gsrc;
#endif
}

DEV void staging_wait() {
#if USE_ASYNC_LDS
  __builtin_amdgcn_s_wait_asynccnt(0);
#endif
}

// fp32 -> bf16 round-to-nearest-even
DEV u16 f2bf(float f) {
  uint32_t u = __float_as_uint(f);
  u += 0x7FFFu + ((u >> 16) & 1u);
  return (u16)(u >> 16);
}

union Frag16 {
  uint4   q[2];
  bf16x16 v;
};

// Load a 16-element bf16 fragment as two 16B chunks from a (16B-aligned) row.
DEV bf16x16 load_frag(const u16* row, int off0, int off1) {
  Frag16 f;
  f.q[0] = *(const uint4*)(row + off0);
  f.q[1] = *(const uint4*)(row + off1);
  return f.v;
}

DEV f32x8 wmma_bf16(bf16x16 a, bf16x16 b, f32x8 c) {
  return __builtin_amdgcn_wmma_f32_16x16x32_bf16(
      /*neg_a=*/false, a, /*neg_b=*/false, b,
      /*c_mod=*/(short)0, c, /*reuse_a=*/false, /*reuse_b=*/false);
}

// ---------------------------------------------------------------------------
// fp32 -> bf16 bulk conversion, 8 elements / thread (fully coalesced).
// ---------------------------------------------------------------------------
__global__ __launch_bounds__(256) void cvt_f32_bf16_kernel(
    const float* __restrict__ in, u16* __restrict__ out, int n8) {
  const int i = blockIdx.x * blockDim.x + threadIdx.x;
  if (i >= n8) return;
  const float4* s = (const float4*)in + (size_t)i * 2;
  const float4 a = s[0], b = s[1];
  union { u16 u[8]; uint4 q; } t;
  t.u[0] = f2bf(a.x); t.u[1] = f2bf(a.y); t.u[2] = f2bf(a.z); t.u[3] = f2bf(a.w);
  t.u[4] = f2bf(b.x); t.u[5] = f2bf(b.y); t.u[6] = f2bf(b.z); t.u[7] = f2bf(b.w);
  ((uint4*)out)[i] = t.q;
}

// ---------------------------------------------------------------------------
// GEMM: Out[m][n] = sum_k A[m][k] * Wt[n][k] + bias[n]      (A, Wt bf16)
// Workgroup tile 128x128, 8 waves in 2x4, each wave 64x32, K-step 64.
// ---------------------------------------------------------------------------
template <bool OUT_BF16>
__global__ __launch_bounds__(256) void wmma_gemm_kernel(
    const u16* __restrict__ A, const u16* __restrict__ Wt,
    const float* __restrict__ bias, void* __restrict__ Out) {
  constexpr int K = 1024, N = 1024, TK = 64, LDT = TK + 8;  // 144B rows
  __shared__ u16 As[128 * LDT];
  __shared__ u16 Bs[128 * LDT];

  const int t  = threadIdx.x;
  const int m0 = blockIdx.x * 128;
  const int n0 = blockIdx.y * 128;
  const int wave = t >> 5, lane = t & 31;
  const int wm = wave >> 2, wn = wave & 3;     // 2 x 4 wave grid
  const int hi = lane >> 4, lw = lane & 15;

  const int sr = t >> 1;           // staging row 0..127 (2 threads / row)
  const int sc = (t & 1) * 32;     // staging col 0 or 32

  const u16* asrc = A  + (size_t)(m0 + sr) * K + sc;
  const u16* bsrc = Wt + (size_t)(n0 + sr) * K + sc;
  u16* adst = &As[sr * LDT + sc];
  u16* bdst = &Bs[sr * LDT + sc];

  f32x8 acc[4][2] = {};

  for (int k0 = 0; k0 < K; k0 += TK) {
    // ---- stage A / W tiles (128 x 64 bf16 each) ----
#pragma unroll
    for (int c = 0; c < 4; ++c) copy16_to_lds(adst + c * 8, asrc + k0 + c * 8);
#pragma unroll
    for (int c = 0; c < 4; ++c) copy16_to_lds(bdst + c * 8, bsrc + k0 + c * 8);
    if (k0 + TK < K) {
      __builtin_prefetch(asrc + k0 + TK, 0, 0);
      __builtin_prefetch(bsrc + k0 + TK, 0, 0);
    }
    staging_wait();
    __syncthreads();

    // ---- 16 WMMAs per wave per K-step ----
#pragma unroll
    for (int kk = 0; kk < TK; kk += 32) {
      bf16x16 af[4], bf[2];
#pragma unroll
      for (int mt = 0; mt < 4; ++mt)
        af[mt] = load_frag(&As[(wm * 64 + mt * 16 + lw) * LDT],
                           kk + 8 * hi, kk + 16 + 8 * hi);
#pragma unroll
      for (int nt = 0; nt < 2; ++nt)
        bf[nt] = load_frag(&Bs[(wn * 32 + nt * 16 + lw) * LDT],
                           kk + 8 * hi, kk + 16 + 8 * hi);
#pragma unroll
      for (int mt = 0; mt < 4; ++mt)
#pragma unroll
        for (int nt = 0; nt < 2; ++nt)
          acc[mt][nt] = wmma_bf16(af[mt], bf[nt], acc[mt][nt]);
    }
    __syncthreads();
  }

  // ---- epilogue: bias add, store (C layout: n = lw, m = r + 8*hi) ----
#pragma unroll
  for (int nt = 0; nt < 2; ++nt) {
    const int gn = n0 + wn * 32 + nt * 16 + lw;
    const float bv = bias[gn];
#pragma unroll
    for (int mt = 0; mt < 4; ++mt) {
#pragma unroll
      for (int r = 0; r < 8; ++r) {
        const int gm = m0 + wm * 64 + mt * 16 + r + 8 * hi;
        const float val = acc[mt][nt][r] + bv;
        if (OUT_BF16)
          ((u16*)Out)[(size_t)gm * N + gn] = f2bf(val);
        else
          ((float*)Out)[(size_t)gm * N + gn] = val;
      }
    }
  }
}

// ---------------------------------------------------------------------------
// Flash attention: one workgroup = one (b, h, 64-query tile).
// 4 waves, each owns 16 query rows. KV tiles of 64, online softmax.
// Qp/Kp/Vp/Ctx are bf16 [B*S x D] with per-head column slice at h*64.
// ---------------------------------------------------------------------------
__global__ __launch_bounds__(128) void flash_attn_kernel(
    const u16* __restrict__ Qp, const u16* __restrict__ Kp,
    const u16* __restrict__ Vp, u16* __restrict__ Ctx) {
  constexpr int LDK = DEP + 8;   // 72 elems -> 144B rows (16B aligned)
  constexpr int LDV = 64 + 8;
  constexpr int LDP = 64 + 8;
  __shared__ u16 Ks[64 * LDK];   // [key][d]
  __shared__ u16 Vt[DEP * LDV];  // [d][key]   (transposed V tile)
  __shared__ u16 Ps[64 * LDP];   // [qrow][key]

  const int bh = blockIdx.x;
  const int b = bh >> 4, h = bh & 15;
  const int q0 = blockIdx.y * 64;
  const int t = threadIdx.x;
  const int wave = t >> 5, lane = t & 31;
  const int hi = lane >> 4, lw = lane & 15;

  // ---- Q fragments for this wave's 16 rows (A layout, direct from global) --
  const u16* qrow =
      Qp + (size_t)(b * S + q0 + wave * 16 + lw) * D + h * DEP;
  const bf16x16 qf0 = load_frag(qrow, 8 * hi, 16 + 8 * hi);       // d 0..31
  const bf16x16 qf1 = load_frag(qrow, 32 + 8 * hi, 48 + 8 * hi);  // d 32..63

  f32x8 O[4] = {};
  float mrow[8], lrow[8];
#pragma unroll
  for (int r = 0; r < 8; ++r) { mrow[r] = -1e30f; lrow[r] = 0.0f; }

  const int sr = t >> 1;           // staging row 0..63
  const int sc = (t & 1) * 32;     // staging col 0 or 32

  for (int kv0 = 0; kv0 < S; kv0 += 64) {
    // ---- stage K tile [key][d] (async path if available) ----
    {
      const u16* src = Kp + (size_t)(b * S + kv0 + sr) * D + h * DEP + sc;
      u16* dst = &Ks[sr * LDK + sc];
#pragma unroll
      for (int c = 0; c < 4; ++c) copy16_to_lds(dst + c * 8, src + c * 8);
    }
    // ---- stage V tile transposed: Vt[d][key] ----
    {
      const u16* src = Vp + (size_t)(b * S + kv0 + sr) * D + h * DEP + sc;
#pragma unroll
      for (int c = 0; c < 4; ++c) {
        union { uint4 q; u16 u[8]; } vv;
        vv.q = *(const uint4*)(src + c * 8);
#pragma unroll
        for (int j = 0; j < 8; ++j)
          Vt[(sc + c * 8 + j) * LDV + sr] = vv.u[j];
      }
    }
    staging_wait();
    __syncthreads();

    // ---- S = Q K^T (per wave: 16 rows x 64 keys) ----
    f32x8 s[4] = {};
#pragma unroll
    for (int nt = 0; nt < 4; ++nt) {
      const u16* krow = &Ks[(nt * 16 + lw) * LDK];
      bf16x16 kb0 = load_frag(krow, 8 * hi, 16 + 8 * hi);            // d 0..31
      bf16x16 kb1 = load_frag(krow, 32 + 8 * hi, 48 + 8 * hi);  // d 32..63
      s[nt] = wmma_bf16(qf0, kb0, s[nt]);
      s[nt] = wmma_bf16(qf1, kb1, s[nt]);
    }

    // ---- online softmax (scale = 1/sqrt(64) = 0.125) ----
#pragma unroll
    for (int r = 0; r < 8; ++r) {
      float s0 = s[0][r] * 0.125f, s1 = s[1][r] * 0.125f;
      float s2 = s[2][r] * 0.125f, s3 = s[3][r] * 0.125f;
      float mx = fmaxf(fmaxf(s0, s1), fmaxf(s2, s3));
      mx = fmaxf(mx, __shfl_xor(mx, 1, 32));
      mx = fmaxf(mx, __shfl_xor(mx, 2, 32));
      mx = fmaxf(mx, __shfl_xor(mx, 4, 32));
      mx = fmaxf(mx, __shfl_xor(mx, 8, 32));
      const float mnew = fmaxf(mrow[r], mx);
      const float alpha = __expf(mrow[r] - mnew);
      mrow[r] = mnew;
      const float p0 = __expf(s0 - mnew);
      const float p1 = __expf(s1 - mnew);
      const float p2 = __expf(s2 - mnew);
      const float p3 = __expf(s3 - mnew);
      s[0][r] = p0; s[1][r] = p1; s[2][r] = p2; s[3][r] = p3;
      float ps = p0 + p1 + p2 + p3;
      ps += __shfl_xor(ps, 1, 32);
      ps += __shfl_xor(ps, 2, 32);
      ps += __shfl_xor(ps, 4, 32);
      ps += __shfl_xor(ps, 8, 32);
      lrow[r] = lrow[r] * alpha + ps;
#pragma unroll
      for (int nt = 0; nt < 4; ++nt) O[nt][r] *= alpha;
    }

    // ---- P -> LDS (own wave rows only; intra-wave dep, no barrier) ----
#pragma unroll
    for (int nt = 0; nt < 4; ++nt)
#pragma unroll
      for (int r = 0; r < 8; ++r)
        Ps[(wave * 16 + r + 8 * hi) * LDP + nt * 16 + lw] = f2bf(s[nt][r]);

    // ---- O += P V ----
    const u16* prow = &Ps[(wave * 16 + lw) * LDP];
    bf16x16 pa0 = load_frag(prow, 8 * hi, 16 + 8 * hi);        // key 0..31
    bf16x16 pa1 = load_frag(prow, 32 + 8 * hi, 48 + 8 * hi);   // key 32..63
#pragma unroll
    for (int nt = 0; nt < 4; ++nt) {
      const u16* vrow = &Vt[(nt * 16 + lw) * LDV];
      bf16x16 vb0 = load_frag(vrow, 8 * hi, 16 + 8 * hi);
      bf16x16 vb1 = load_frag(vrow, 32 + 8 * hi, 48 + 8 * hi);
      O[nt] = wmma_bf16(pa0, vb0, O[nt]);
      O[nt] = wmma_bf16(pa1, vb1, O[nt]);
    }
    __syncthreads();  // protect Ks/Vt before next staging
  }

  // ---- normalize + write ctx (merged heads, bf16) ----
#pragma unroll
  for (int nt = 0; nt < 4; ++nt) {
#pragma unroll
    for (int r = 0; r < 8; ++r) {
      const float val = O[nt][r] / lrow[r];
      const size_t row = (size_t)(b * S + q0 + wave * 16 + r + 8 * hi);
      Ctx[row * D + h * DEP + nt * 16 + lw] = f2bf(val);
    }
  }
}

// ---------------------------------------------------------------------------
// Host launcher.
// d_in order: v, k, q, wq_w, wq_b, wk_w, wk_b, wv_w, wv_b, dense_w, dense_b
// ---------------------------------------------------------------------------
extern "C" void kernel_launch(void* const* d_in, const int* in_sizes, int n_in,
                              void* d_out, int out_size, void* d_ws,
                              size_t ws_size, hipStream_t stream) {
  (void)in_sizes; (void)n_in; (void)out_size; (void)ws_size;

  const float* v   = (const float*)d_in[0];
  const float* k   = (const float*)d_in[1];
  const float* q   = (const float*)d_in[2];
  const float* wq  = (const float*)d_in[3];
  const float* wqb = (const float*)d_in[4];
  const float* wk  = (const float*)d_in[5];
  const float* wkb = (const float*)d_in[6];
  const float* wv  = (const float*)d_in[7];
  const float* wvb = (const float*)d_in[8];
  const float* wd  = (const float*)d_in[9];
  const float* wdb = (const float*)d_in[10];
  float* out = (float*)d_out;

  // Workspace layout (bf16 buffers), total 120 MB:
  const size_t ACT = (size_t)M_TOK * D;   // 8 Mi elems
  const size_t WGT = (size_t)D * D;       // 1 Mi elems
  u16* qb  = (u16*)d_ws;                  // converted activations
  u16* kb  = qb + ACT;
  u16* vb  = kb + ACT;
  u16* wqc = vb + ACT;                    // converted weights
  u16* wkc = wqc + WGT;
  u16* wvc = wkc + WGT;
  u16* wdc = wvc + WGT;
  u16* Qp  = wdc + WGT;                   // projected Q/K/V
  u16* Kp  = Qp + ACT;
  u16* Vp  = Kp + ACT;
  u16* Ctx = Vp + ACT;                    // merged-head context

  // ---- one-shot bf16 conversion ----
  const int actN8 = (int)(ACT / 8), wgtN8 = (int)(WGT / 8);
  cvt_f32_bf16_kernel<<<(actN8 + 255) / 256, 256, 0, stream>>>(q, qb, actN8);
  cvt_f32_bf16_kernel<<<(actN8 + 255) / 256, 256, 0, stream>>>(k, kb, actN8);
  cvt_f32_bf16_kernel<<<(actN8 + 255) / 256, 256, 0, stream>>>(v, vb, actN8);
  cvt_f32_bf16_kernel<<<(wgtN8 + 255) / 256, 256, 0, stream>>>(wq, wqc, wgtN8);
  cvt_f32_bf16_kernel<<<(wgtN8 + 255) / 256, 256, 0, stream>>>(wk, wkc, wgtN8);
  cvt_f32_bf16_kernel<<<(wgtN8 + 255) / 256, 256, 0, stream>>>(wv, wvc, wgtN8);
  cvt_f32_bf16_kernel<<<(wgtN8 + 255) / 256, 256, 0, stream>>>(wd, wdc, wgtN8);

  const dim3 ggrid(M_TOK / 128, D / 128);  // 64 x 8
  const dim3 gblock(256);

  // ---- projections (bf16 in, bf16 out) ----
  wmma_gemm_kernel<true><<<ggrid, gblock, 0, stream>>>(qb, wqc, wqb, Qp);
  wmma_gemm_kernel<true><<<ggrid, gblock, 0, stream>>>(kb, wkc, wkb, Kp);
  wmma_gemm_kernel<true><<<ggrid, gblock, 0, stream>>>(vb, wvc, wvb, Vp);

  // ---- attention: grid = (B*H, S/64) = (64, 32), 128 threads ----
  flash_attn_kernel<<<dim3(64, 32), dim3(128), 0, stream>>>(Qp, Kp, Vp, Ctx);

  // ---- output dense (bf16 in, fp32 out) ----
  wmma_gemm_kernel<false><<<ggrid, gblock, 0, stream>>>(Ctx, wdc, wdb, out);
}
